// TritonLinearAttention2_42047729827882
// MI455X (gfx1250) — matrix operand order repaired and checked by reading, not compiled, not run on hardware
//
#include <hip/hip_runtime.h>
#include <hip/hip_bf16.h>

typedef __bf16 bf16_t;
typedef __attribute__((ext_vector_type(16))) __bf16 v16bf;
typedef __attribute__((ext_vector_type(8)))  __bf16 v8bf;
typedef __attribute__((ext_vector_type(8)))  float  v8f;

#define N_SPATIAL 3136   // 56*56
#define C_TOT     256
#define BATCH     32
#define HEADS     8
#define HD        32

// ---------------- WMMA fragment layouts (CDNA5 ISA) ----------------
// A 16x32 bf16: lanes 0-15: M=lane, elems 0..7->K 0..7,  8..15->K 16..23
//               lanes16-31: M=l-16, elems 0..7->K 8..15, 8..15->K 24..31
//   => 8-elem run r (K = r*8..r*8+7): lane = (m&15)|((r&1)<<4), elemoff = (r>=2)?8:0
// B 32x16 bf16: N=lane&15; K1/K2 stage B in A's K order. K3's q/kv use lanes 0-15: K 0..15, 16-31: K 16..31
//   => K3 8-elem run r: lane = (col&15)|((r>=2)?16:0), elemoff = (r&1)*8

__device__ inline v16bf load_frag(const bf16_t* tile, int lane) {
  return ((const v16bf*)tile)[lane];
}

__device__ inline v8f wmma_bf16(v16bf a, v16bf b, v8f c) {
  return __builtin_amdgcn_wmma_f32_16x16x32_bf16(false, a, false, b, (short)0, c,
                                                 false, false);
}

__device__ inline v8bf cvt8(float4 a, float4 b) {
  v8bf r;
  r[0] = (bf16_t)a.x; r[1] = (bf16_t)a.y; r[2] = (bf16_t)a.z; r[3] = (bf16_t)a.w;
  r[4] = (bf16_t)b.x; r[5] = (bf16_t)b.y; r[6] = (bf16_t)b.z; r[7] = (bf16_t)b.w;
  return r;
}

// ---------------- K1: grouped 1x1 conv (q/k projection) + ELU+1 ----------------
// grid (49, 4, 32): x = n-block (64 cols), y = {g, mhalf}, z = batch. 256 thr.
// q is stored in K3's B-fragment global order; k stays row-major [c][n] for K2.
__global__ __launch_bounds__(256) void qk_gemm_kernel(
    const float* __restrict__ x, const float* __restrict__ qkw,
    const float* __restrict__ qkb, bf16_t* __restrict__ qfrag,
    bf16_t* __restrict__ kb) {
  const int nblk = blockIdx.x;
  const int g = blockIdx.y >> 1, mhalf = blockIdx.y & 1;
  const int b = blockIdx.z;
  const int t = threadIdx.x, wave = t >> 5, lane = t & 31;
  const int n0 = nblk * 64;

  __shared__ __align__(32) bf16_t As[8 * 512];  // A-frag order, 8 m-subtiles
  __shared__ __align__(32) bf16_t Bs[4 * 512];  // B-frag order, 4 n-subtiles

  const float* W = qkw + (size_t)(g * 256 + mhalf * 128) * 128;   // [128 m][128 k]
  const float* X = x + ((size_t)b * C_TOT + g * 128) * N_SPATIAL; // [128 k][3136]

  v8f acc[4];
#pragma unroll
  for (int nt = 0; nt < 4; ++nt) acc[nt] = (v8f)(0.0f);

  // staging assignments (fixed per thread)
  const int wm = t >> 1;                 // W row (0..127)
  const int xcol = t & 63, xr = t >> 6;  // X col (0..63), k-run (0..3)
  bf16_t* const xdst = &Bs[(xcol >> 4) * 512 +
                           (((xcol & 15) | ((xr & 1) << 4)) << 4) +
                           ((xr >= 2) ? 8 : 0)];

  for (int kt = 0; kt < 4; ++kt) {
    const int k0 = kt * 32;
    __syncthreads();
    {  // stage W tile 128x32 -> A-frag LDS: 2 vectorized 8-runs per thread
#pragma unroll
      for (int s = 0; s < 2; ++s) {
        int r = (t & 1) * 2 + s;
        const float* wp = W + (size_t)wm * 128 + k0 + r * 8;
        float4 lo = ((const float4*)wp)[0], hi = ((const float4*)wp)[1];
        bf16_t* dst = &As[(wm >> 4) * 512 +
                          (((wm & 15) | ((r & 1) << 4)) << 4) +
                          ((r >= 2) ? 8 : 0)];
        *(v8bf*)dst = cvt8(lo, hi);
      }
    }
    {  // stage X tile 32x64 -> B-frag LDS: column-oriented, wave-coalesced loads
      const float* xp = X + (size_t)(k0 + xr * 8) * N_SPATIAL + n0 + xcol;
      v8bf xv;
#pragma unroll
      for (int j = 0; j < 8; ++j) xv[j] = (bf16_t)xp[(size_t)j * N_SPATIAL];
      *(v8bf*)xdst = xv;
      if (kt < 3)  // prefetch next K-slab (global_prefetch_b8)
        __builtin_prefetch(xp + 32 * N_SPATIAL, 0, 1);
    }
    __syncthreads();
    v16bf a = load_frag(&As[wave * 512], lane);
#pragma unroll
    for (int nt = 0; nt < 4; ++nt) {
      v16bf bf = load_frag(&Bs[nt * 512], lane);
      acc[nt] = wmma_bf16(a, bf, acc[nt]);
    }
  }

  // ---- epilogue: bias + elu(v)+1, store bf16 ----
  const int m0 = mhalf * 128 + wave * 16 + (lane >> 4) * 8;  // first channel of run
  float bias[8];
#pragma unroll
  for (int i = 0; i < 8; ++i) bias[i] = qkb[g * 256 + m0 + i];

  if (g == 0) {
    // q -> B-fragment global order: 8 consecutive i == 16B contiguous run
    const int h = m0 >> 5, d0 = m0 & 31, col = lane & 15;
    const size_t base =
        ((((size_t)(b * HEADS + h) * 49 + nblk) * 4) << 9) +
        ((col | ((d0 & 16) ? 16 : 0)) << 4) + (d0 & 15);
#pragma unroll
    for (int nt = 0; nt < 4; ++nt) {
      v8bf pk;
#pragma unroll
      for (int i = 0; i < 8; ++i) {
        float v = acc[nt][i] + bias[i];
        pk[i] = (bf16_t)((v > 0.0f) ? (v + 1.0f) : __expf(v));
      }
      *(v8bf*)(qfrag + base + ((size_t)nt << 9)) = pk;  // 16B store
    }
  } else {
    bf16_t* dst = kb + ((size_t)b * C_TOT + m0) * N_SPATIAL + n0 + (lane & 15);
#pragma unroll
    for (int i = 0; i < 8; ++i) {
#pragma unroll
      for (int nt = 0; nt < 4; ++nt) {
        float v = acc[nt][i] + bias[i];
        v = (v > 0.0f) ? (v + 1.0f) : __expf(v);
        dst[(size_t)i * N_SPATIAL + nt * 16] = (bf16_t)v;
      }
    }
  }
}

// ---------------- K2: kv[d][e] = sum_n k[d][n]*v[e][n]; ksum[d] ----------------
// grid (256): one block per (b,h). 128 threads = 4 waves; wave = (mt,nt) tile.
__global__ __launch_bounds__(128) void kv_kernel(
    const bf16_t* __restrict__ kb, const float* __restrict__ x,
    bf16_t* __restrict__ kvA, float* __restrict__ ksum) {
  const int bh = blockIdx.x;
  const int b = bh >> 3, h = bh & 7;
  const int t = threadIdx.x, wave = t >> 5, lane = t & 31;
  const int mt = wave >> 1, nt = wave & 1;

  const bf16_t* K = kb + ((size_t)b * C_TOT + h * HD) * N_SPATIAL;
  const float*  V = x  + ((size_t)b * C_TOT + h * HD) * N_SPATIAL;

  __shared__ __align__(32) bf16_t Ksf[2 * 512];  // A-frag order
  __shared__ __align__(32) bf16_t Vsf[2 * 512];  // B-frag order
  __shared__ float red[128];

  // staging: thread owns (row, 8-elem k-run) -> one 16B LDS store each
  const int row = t >> 2, r = t & 3;
  bf16_t* const kdst = &Ksf[(row >> 4) * 512 +
                            (((row & 15) | ((r & 1) << 4)) << 4) +
                            ((r >= 2) ? 8 : 0)];                    // A-frag (m=d)
  bf16_t* const vdst = &Vsf[(row >> 4) * 512 +
                            (((row & 15) | ((r & 1) << 4)) << 4) +
                            ((r >= 2) ? 8 : 0)];                    // B-frag (col=e)
  const bf16_t* const ksrc0 = K + (size_t)row * N_SPATIAL + r * 8;
  const float*  const vsrc0 = V + (size_t)row * N_SPATIAL + r * 8;

  float ks_part = 0.0f;
  v8f acc = (v8f)(0.0f);
  for (int kc = 0; kc < N_SPATIAL / 32; ++kc) {
    const int nb = kc * 32;
    __syncthreads();
    {
      v8bf kv8 = *(const v8bf*)(ksrc0 + nb);  // pure 16B bf16 copy
      *(v8bf*)kdst = kv8;
#pragma unroll
      for (int j = 0; j < 8; ++j) ks_part += (float)kv8[j];
      float4 lo = ((const float4*)(vsrc0 + nb))[0];
      float4 hi = ((const float4*)(vsrc0 + nb))[1];
      *(v8bf*)vdst = cvt8(lo, hi);
    }
    __syncthreads();
    v16bf a  = load_frag(&Ksf[mt * 512], lane);
    v16bf bf = load_frag(&Vsf[nt * 512], lane);
    acc = wmma_bf16(a, bf, acc);
  }

  // store kv in K3 A-fragment order: subtile = e>>4 (= nt), element (m=e&15, k=d)
  bf16_t* outp = kvA + (size_t)bh * (HD * HD);
#pragma unroll
  for (int i = 0; i < 8; ++i) {
    int d = mt * 16 + (lane >> 4) * 8 + i;
    int e = nt * 16 + (lane & 15);
    int fi = (((e & 15) | (((d >> 4) & 1) << 4)) << 4) |
             (d & 15);
    outp[nt * 512 + fi] = (bf16_t)acc[i];
  }

  // ksum[d]: reduce 4 partials per d (threads t = d*4 + r)
  red[t] = ks_part;
  __syncthreads();
  if (t < 32)
    ksum[bh * HD + t] =
        red[t * 4] + red[t * 4 + 1] + red[t * 4 + 2] + red[t * 4 + 3];
}

// ---------------- K3: num + denom + normalize + depthwise 3x3 pe ----------------
// grid (49, 8, 32): x = n-block (64), y = head, z = batch. 128 thr = 4 waves.
// q and kv are already in fragment order in global memory -> no LDS staging.
__global__ __launch_bounds__(128) void out_kernel(
    const bf16_t* __restrict__ qfrag, const float* __restrict__ x,
    const bf16_t* __restrict__ kvA, const float* __restrict__ ksum,
    const float* __restrict__ pew, const float* __restrict__ peb,
    float* __restrict__ out) {
  const int nblk = blockIdx.x, h = blockIdx.y, b = blockIdx.z;
  const int bh = b * HEADS + h;
  const int t = threadIdx.x, wave = t >> 5, lane = t & 31;
  const int n0 = nblk * 64;

  __shared__ float ks_s[32];
  if (t < 32) ks_s[t] = ksum[bh * HD + t];
  __syncthreads();

  // fragments straight from global (2x global_load_b128 each)
  const bf16_t* A0 = kvA + (size_t)bh * (HD * HD);
  v16bf a0 = load_frag(A0, lane);
  v16bf a1 = load_frag(A0 + 512, lane);
  const bf16_t* Qf = qfrag + ((((size_t)bh * 49 + nblk) * 4 + wave) << 9);
  v16bf bf = load_frag(Qf, lane);

  v8f acc[2];
  acc[0] = wmma_bf16(a0, bf, (v8f)(0.0f));
  acc[1] = wmma_bf16(a1, bf, (v8f)(0.0f));

  // denom from the B fragment itself: lane holds q[d=half*16+e][n=col]
  const int half = lane >> 4;
  float part = 0.0f;
#pragma unroll
  for (int e = 0; e < 16; ++e) part += (float)bf[e] * ks_s[half * 16 + e];
  const float den = part + __shfl_xor(part, 16) + (float)N_SPATIAL * 1e-6f;
  const float rden = 1.0f / den;  // num/n/(raw/n+1e-6) == num/(raw + n*1e-6)

  // hoist pe-conv geometry: identical for all 16 outputs of this lane
  const int n = n0 + wave * 16 + (lane & 15);
  const int yy = n / 56, xx = n % 56;
  int   off[9];
  float msk[9];
#pragma unroll
  for (int dy = -1; dy <= 1; ++dy) {
#pragma unroll
    for (int dx = -1; dx <= 1; ++dx) {
      int tap = (dy + 1) * 3 + (dx + 1);
      int y2 = yy + dy, x2 = xx + dx;
      bool v = (y2 >= 0) && (y2 < 56) && (x2 >= 0) && (x2 < 56);
      int y2c = min(max(y2, 0), 55), x2c = min(max(x2, 0), 55);
      off[tap] = y2c * 56 + x2c;
      msk[tap] = v ? 1.0f : 0.0f;
    }
  }

  const int e0 = (lane >> 4) * 8;
  const float* xb = x + (size_t)b * C_TOT * N_SPATIAL;
#pragma unroll
  for (int mtile = 0; mtile < 2; ++mtile) {
#pragma unroll
    for (int i = 0; i < 8; ++i) {
      int c = h * HD + mtile * 16 + e0 + i;
      float r = acc[mtile][i] * rden;

      const float* xc = xb + (size_t)c * N_SPATIAL;
      const float* wc = pew + c * 9;
      float pe = peb[c];
#pragma unroll
      for (int tap = 0; tap < 9; ++tap)
        pe += xc[off[tap]] * (msk[tap] * wc[tap]);

      out[(size_t)c * N_SPATIAL + (size_t)b * C_TOT * N_SPATIAL + n] = r + pe;
    }
  }
}

// ---------------- launch ----------------
extern "C" void kernel_launch(void* const* d_in, const int* in_sizes, int n_in,
                              void* d_out, int out_size, void* d_ws,
                              size_t ws_size, hipStream_t stream) {
  const float* x    = (const float*)d_in[0];
  const float* qk_w = (const float*)d_in[1];
  const float* qk_b = (const float*)d_in[2];
  const float* pe_w = (const float*)d_in[3];
  const float* pe_b = (const float*)d_in[4];
  float* out = (float*)d_out;

  const size_t QK_ELEMS = (size_t)BATCH * C_TOT * N_SPATIAL;  // 25.7M
  bf16_t* qfrag = (bf16_t*)d_ws;
  bf16_t* kb    = qfrag + QK_ELEMS;
  bf16_t* kvA   = kb + QK_ELEMS;                      // 256 * 1024 elems
  float*  ksum  = (float*)(kvA + (size_t)256 * HD * HD);

  qk_gemm_kernel<<<dim3(49, 4, BATCH), 256, 0, stream>>>(x, qk_w, qk_b, qfrag, kb);
  kv_kernel<<<dim3(BATCH * HEADS), 128, 0, stream>>>(kb, x, kvA, ksum);
  out_kernel<<<dim3(49, HEADS, BATCH), 128, 0, stream>>>(qfrag, x, kvA, ksum,
                                                         pe_w, pe_b, out);
  (void)in_sizes; (void)n_in; (void)out_size; (void)ws_size;
}
